// Attention_55336358642892
// MI455X (gfx1250) — compile-verified, matrix operand order checked by reading, not run on hardware
//
#include <hip/hip_runtime.h>
#include <hip/hip_bf16.h>

typedef __attribute__((ext_vector_type(16))) _Float16 v16h;
typedef __attribute__((ext_vector_type(8)))  float    v8f;

#define EMBED    1024
#define HEADS    16
#define HEAD_DIM 64
#define NSEQ     2048
#define BATCH    2
#define MROWS    (BATCH * NSEQ)        // 4096
#define SCALE_F  0.125f                // HEAD_DIM^-0.5
#define LN_EPS   1e-5f
#define NEG_BIG  (-1e30f)

static __device__ __forceinline__ v8f wmma_f16(v16h a, v16h b, v8f c) {
  // D = A(16x32 f16) * B(32x16 f16) + C(16x16 f32)
  return __builtin_amdgcn_wmma_f32_16x16x32_f16(
      /*neg_a=*/false, a, /*neg_b=*/false, b,
      /*c_mod=*/(short)0, c, /*reuse_a=*/false, /*reuse_b=*/false);
}

// In-order LDS pipe within a wave: wait DS ops then fence the compiler.
static __device__ __forceinline__ void lds_fence_wave() {
  asm volatile("s_wait_dscnt 0x0" ::: "memory");
}

// ---------------------------------------------------------------------------
// fp32 -> f16 elementwise (weight conversion)
// ---------------------------------------------------------------------------
__global__ void cvt_f32_f16(const float* __restrict__ s, _Float16* __restrict__ d, int n) {
  int i = blockIdx.x * 256 + threadIdx.x;
  if (i < n) d[i] = (_Float16)s[i];
}

// ---------------------------------------------------------------------------
// LayerNorm: one wave per row of 1024, fp32 in, f16 out
// ---------------------------------------------------------------------------
__global__ __launch_bounds__(32) void ln_kernel(const float* __restrict__ x,
                                                const float* __restrict__ w,
                                                const float* __restrict__ bns,
                                                _Float16* __restrict__ xn) {
  int row  = blockIdx.x;
  int lane = threadIdx.x;
  const float4* xr = (const float4*)(x + (size_t)row * EMBED);
  float4 v[8];
  float s = 0.f, s2 = 0.f;
#pragma unroll
  for (int i = 0; i < 8; ++i) {
    v[i] = xr[lane + 32 * i];
    s  += v[i].x + v[i].y + v[i].z + v[i].w;
    s2 += v[i].x * v[i].x + v[i].y * v[i].y + v[i].z * v[i].z + v[i].w * v[i].w;
  }
#pragma unroll
  for (int off = 16; off >= 1; off >>= 1) {
    s  += __shfl_xor(s, off, 32);
    s2 += __shfl_xor(s2, off, 32);
  }
  float mu  = s * (1.f / EMBED);
  float var = s2 * (1.f / EMBED) - mu * mu;
  float rs  = rsqrtf(var + LN_EPS);
#pragma unroll
  for (int i = 0; i < 8; ++i) {
    int idx = (lane + 32 * i) * 4;
    float4 wv = ((const float4*)w)[lane + 32 * i];
    float4 bv = ((const float4*)bns)[lane + 32 * i];
    _Float16* o = xn + (size_t)row * EMBED + idx;
    o[0] = (_Float16)((v[i].x - mu) * rs * wv.x + bv.x);
    o[1] = (_Float16)((v[i].y - mu) * rs * wv.y + bv.y);
    o[2] = (_Float16)((v[i].z - mu) * rs * wv.z + bv.z);
    o[3] = (_Float16)((v[i].w - mu) * rs * wv.w + bv.w);
  }
}

// ---------------------------------------------------------------------------
// C = A(M x K) @ W^T, both f16, K innermost. 4 waves per block share one
// 64-wide W strip (WGP$ reuse); each wave computes a 16x64 output strip with
// a manually double-buffered K-loop (preload kb+1 while computing kb).
// mode 0: dst[m][col] f16 row-major
// mode 1: dst = [B,H,N,D] (Q/K head layout)
// mode 2: dst = [B,H,D,N] (V transposed for flash-attention B fragments)
// ---------------------------------------------------------------------------
__global__ __launch_bounds__(128) void gemm_qkv(const _Float16* __restrict__ A,
                                                const _Float16* __restrict__ W,
                                                _Float16* __restrict__ dst, int mode) {
  int lane = threadIdx.x & 31;
  int wave = threadIdx.x >> 5;
  int hi = lane >> 4, ln16 = lane & 15;
  int bx = blockIdx.x;                 // (4096/64) * (1024/64) = 64*16 blocks
  int nt = bx & 15;                    // 64-wide N strip (shared by 4 waves)
  int mt = (bx >> 4) * 4 + wave;       // per-wave M tile
  int m0 = mt * 16, ncol0 = nt * 64;
  const _Float16* arow = A + (size_t)(m0 + ln16) * EMBED + hi * 16;
  const _Float16* wrow = W + (size_t)(ncol0 + ln16) * EMBED + hi * 16;
  v8f acc0 = {}, acc1 = {}, acc2 = {}, acc3 = {};

  v16h a  = *(const v16h*)(arow);
  v16h b0 = *(const v16h*)(wrow);
  v16h b1 = *(const v16h*)(wrow + 16 * EMBED);
  v16h b2 = *(const v16h*)(wrow + 32 * EMBED);
  v16h b3 = *(const v16h*)(wrow + 48 * EMBED);
  for (int kb = 0; kb < EMBED / 32 - 1; ++kb) {
    const _Float16* apn = arow + (kb + 1) * 32;
    const _Float16* wpn = wrow + (kb + 1) * 32;
    v16h an  = *(const v16h*)(apn);
    v16h b0n = *(const v16h*)(wpn);
    v16h b1n = *(const v16h*)(wpn + 16 * EMBED);
    v16h b2n = *(const v16h*)(wpn + 32 * EMBED);
    v16h b3n = *(const v16h*)(wpn + 48 * EMBED);
    __builtin_prefetch(apn + 256, 0, 3);
    acc0 = wmma_f16(a, b0, acc0);
    acc1 = wmma_f16(a, b1, acc1);
    acc2 = wmma_f16(a, b2, acc2);
    acc3 = wmma_f16(a, b3, acc3);
    a = an; b0 = b0n; b1 = b1n; b2 = b2n; b3 = b3n;
  }
  acc0 = wmma_f16(a, b0, acc0);
  acc1 = wmma_f16(a, b1, acc1);
  acc2 = wmma_f16(a, b2, acc2);
  acc3 = wmma_f16(a, b3, acc3);

  float accs[4][8];
#pragma unroll
  for (int r = 0; r < 8; ++r) {
    accs[0][r] = acc0[r]; accs[1][r] = acc1[r];
    accs[2][r] = acc2[r]; accs[3][r] = acc3[r];
  }
#pragma unroll
  for (int j = 0; j < 4; ++j) {
    int col = ncol0 + j * 16 + ln16;
    if (mode == 0) {
#pragma unroll
      for (int r = 0; r < 8; ++r) {
        int m = m0 + r + 8 * hi;
        dst[(size_t)m * EMBED + col] = (_Float16)accs[j][r];
      }
    } else {
      int h = col >> 6, d = col & 63;
#pragma unroll
      for (int r = 0; r < 8; ++r) {
        int m  = m0 + r + 8 * hi;
        int b2 = m >> 11;               // batch
        int n  = m & (NSEQ - 1);        // seq pos
        size_t base = (size_t)(b2 * HEADS + h);
        if (mode == 1)
          dst[(base * NSEQ + n) * HEAD_DIM + d] = (_Float16)accs[j][r];
        else
          dst[(base * HEAD_DIM + d) * NSEQ + n] = (_Float16)accs[j][r];
      }
    }
  }
}

// ---------------------------------------------------------------------------
// Flash attention: 4 waves per block handle 4 adjacent 16-query tiles of the
// same (b,h) so the K/V stream is shared through WGP$. Scores stay in
// registers; online softmax via width-16 xor reductions. P re-layout C->A via
// per-wave LDS (in-order DS pipe + s_wait_dscnt, no block barrier needed).
// ---------------------------------------------------------------------------
__global__ __launch_bounds__(128) void flash_attn(const _Float16* __restrict__ q,
                                                  const _Float16* __restrict__ k,
                                                  const _Float16* __restrict__ vt,
                                                  _Float16* __restrict__ ao) {
  __shared__ _Float16 pl[4][16 * 32];
  int lane = threadIdx.x & 31;
  int wave = threadIdx.x >> 5;
  int hi = lane >> 4, ln16 = lane & 15;
  int tile = blockIdx.x * 4 + wave;    // B*H*(N/16) tiles, 4 per block
  int qb = tile & 127;
  int h  = (tile >> 7) & (HEADS - 1);
  int b  = tile >> 11;
  int m0 = qb * 16;
  _Float16* plw = &pl[wave][0];
  size_t bh = (size_t)(b * HEADS + h);
  const _Float16* qb_p = q  + bh * NSEQ * HEAD_DIM;
  const _Float16* kb_p = k  + bh * NSEQ * HEAD_DIM;
  const _Float16* vb_p = vt + bh * HEAD_DIM * NSEQ;

  // Q A-fragments for d = 0..31 and d = 32..63 (loaded once)
  const _Float16* qrow = qb_p + (size_t)(m0 + ln16) * HEAD_DIM;
  v16h a0 = *(const v16h*)(qrow + hi * 16);
  v16h a1 = *(const v16h*)(qrow + 32 + hi * 16);

  v8f o0 = {}, o1 = {}, o2 = {}, o3 = {};
  float mrun[8], lrun[8];
#pragma unroll
  for (int r = 0; r < 8; ++r) { mrun[r] = NEG_BIG; lrun[r] = 0.f; }

  int kbmax = (m0 + 15) >> 5;          // causal: last key block touching this tile
  for (int kb = 0; kb <= kbmax; ++kb) {
    int n0 = kb * 32;
    // ---- S = Q @ K^T for 32 keys (two 16x16 C tiles) ----
    v8f s0 = {}, s1 = {};
    const _Float16* kr0 = kb_p + (size_t)(n0 + ln16) * HEAD_DIM;
    const _Float16* kr1 = kr0 + 16 * HEAD_DIM;
    v16h b00 = *(const v16h*)(kr0 + hi * 16);
    v16h b01 = *(const v16h*)(kr0 + 32 + hi * 16);
    v16h b10 = *(const v16h*)(kr1 + hi * 16);
    v16h b11 = *(const v16h*)(kr1 + 32 + hi * 16);
    s0 = wmma_f16(a0, b00, s0); s0 = wmma_f16(a1, b01, s0);
    s1 = wmma_f16(a0, b10, s1); s1 = wmma_f16(a1, b11, s1);

    // ---- scale, causal mask, online softmax ----
#pragma unroll
    for (int r = 0; r < 8; ++r) {
      int rowg = m0 + r + 8 * hi;
      float v0 = s0[r] * SCALE_F; if (n0 + ln16      > rowg) v0 = NEG_BIG;
      float v1 = s1[r] * SCALE_F; if (n0 + 16 + ln16 > rowg) v1 = NEG_BIG;
      float mx = fmaxf(v0, v1);
#pragma unroll
      for (int off = 8; off >= 1; off >>= 1) mx = fmaxf(mx, __shfl_xor(mx, off, 16));
      float mnew  = fmaxf(mrun[r], mx);
      float alpha = __expf(mrun[r] - mnew);
      float p0 = __expf(v0 - mnew);
      float p1 = __expf(v1 - mnew);
      float rsum = p0 + p1;
#pragma unroll
      for (int off = 8; off >= 1; off >>= 1) rsum += __shfl_xor(rsum, off, 16);
      lrun[r] = lrun[r] * alpha + rsum;
      mrun[r] = mnew;
      o0[r] *= alpha; o1[r] *= alpha; o2[r] *= alpha; o3[r] *= alpha;
      s0[r] = p0; s1[r] = p1;
    }

    // ---- P: C-fragment (f32) -> A-fragment (f16) via per-wave LDS ----
#pragma unroll
    for (int r = 0; r < 8; ++r) {
      plw[(r + 8 * hi) * 32 + ln16]      = (_Float16)s0[r];
      plw[(r + 8 * hi) * 32 + 16 + ln16] = (_Float16)s1[r];
    }
    lds_fence_wave();                  // DS pipe is in-order within a wave
    v16h ap = *(const v16h*)(&plw[ln16 * 32 + hi * 16]);
    lds_fence_wave();

    // ---- O += P @ V : V^T layout makes B fragments contiguous ----
    const _Float16* vb0 = vb_p + n0 + hi * 16;     // + dcol*NSEQ below
    v16h bv0 = *(const v16h*)(vb0 + (size_t)(ln16 +  0) * NSEQ);
    v16h bv1 = *(const v16h*)(vb0 + (size_t)(ln16 + 16) * NSEQ);
    v16h bv2 = *(const v16h*)(vb0 + (size_t)(ln16 + 32) * NSEQ);
    v16h bv3 = *(const v16h*)(vb0 + (size_t)(ln16 + 48) * NSEQ);
    o0 = wmma_f16(ap, bv0, o0);
    o1 = wmma_f16(ap, bv1, o1);
    o2 = wmma_f16(ap, bv2, o2);
    o3 = wmma_f16(ap, bv3, o3);
  }

  // ---- normalize and write attention output [B*N, EMBED] f16 ----
#pragma unroll
  for (int r = 0; r < 8; ++r) {
    float inv = 1.f / lrun[r];
    int rowg = m0 + r + 8 * hi;
    _Float16* orow = ao + ((size_t)(b * NSEQ + rowg)) * EMBED + h * HEAD_DIM;
    orow[ 0 + ln16] = (_Float16)(o0[r] * inv);
    orow[16 + ln16] = (_Float16)(o1[r] * inv);
    orow[32 + ln16] = (_Float16)(o2[r] * inv);
    orow[48 + ln16] = (_Float16)(o3[r] * inv);
  }
}

// ---------------------------------------------------------------------------
// out = residual + ao @ Wo^T + bo  (fp32 output); same blocking as gemm_qkv
// ---------------------------------------------------------------------------
__global__ __launch_bounds__(128) void out_proj(const _Float16* __restrict__ A,
                                                const _Float16* __restrict__ W,
                                                const float* __restrict__ x,
                                                const float* __restrict__ bo,
                                                float* __restrict__ out) {
  int lane = threadIdx.x & 31;
  int wave = threadIdx.x >> 5;
  int hi = lane >> 4, ln16 = lane & 15;
  int bx = blockIdx.x;
  int nt = bx & 15;
  int mt = (bx >> 4) * 4 + wave;
  int m0 = mt * 16, ncol0 = nt * 64;
  const _Float16* arow = A + (size_t)(m0 + ln16) * EMBED + hi * 16;
  const _Float16* wrow = W + (size_t)(ncol0 + ln16) * EMBED + hi * 16;
  v8f acc0 = {}, acc1 = {}, acc2 = {}, acc3 = {};

  v16h a  = *(const v16h*)(arow);
  v16h b0 = *(const v16h*)(wrow);
  v16h b1 = *(const v16h*)(wrow + 16 * EMBED);
  v16h b2 = *(const v16h*)(wrow + 32 * EMBED);
  v16h b3 = *(const v16h*)(wrow + 48 * EMBED);
  for (int kb = 0; kb < EMBED / 32 - 1; ++kb) {
    const _Float16* apn = arow + (kb + 1) * 32;
    const _Float16* wpn = wrow + (kb + 1) * 32;
    v16h an  = *(const v16h*)(apn);
    v16h b0n = *(const v16h*)(wpn);
    v16h b1n = *(const v16h*)(wpn + 16 * EMBED);
    v16h b2n = *(const v16h*)(wpn + 32 * EMBED);
    v16h b3n = *(const v16h*)(wpn + 48 * EMBED);
    __builtin_prefetch(apn + 256, 0, 3);
    acc0 = wmma_f16(a, b0, acc0);
    acc1 = wmma_f16(a, b1, acc1);
    acc2 = wmma_f16(a, b2, acc2);
    acc3 = wmma_f16(a, b3, acc3);
    a = an; b0 = b0n; b1 = b1n; b2 = b2n; b3 = b3n;
  }
  acc0 = wmma_f16(a, b0, acc0);
  acc1 = wmma_f16(a, b1, acc1);
  acc2 = wmma_f16(a, b2, acc2);
  acc3 = wmma_f16(a, b3, acc3);

  float accs[4][8];
#pragma unroll
  for (int r = 0; r < 8; ++r) {
    accs[0][r] = acc0[r]; accs[1][r] = acc1[r];
    accs[2][r] = acc2[r]; accs[3][r] = acc3[r];
  }
#pragma unroll
  for (int j = 0; j < 4; ++j) {
    int col = ncol0 + j * 16 + ln16;
    float bias = bo[col];
#pragma unroll
    for (int r = 0; r < 8; ++r) {
      int m = m0 + r + 8 * hi;
      size_t idx = (size_t)m * EMBED + col;
      out[idx] = x[idx] + bias + accs[j][r];
    }
  }
}

// ---------------------------------------------------------------------------
extern "C" void kernel_launch(void* const* d_in, const int* in_sizes, int n_in,
                              void* d_out, int out_size, void* d_ws, size_t ws_size,
                              hipStream_t stream) {
  (void)in_sizes; (void)n_in; (void)out_size; (void)ws_size;
  const float* x    = (const float*)d_in[0];
  const float* ln_w = (const float*)d_in[1];
  const float* ln_b = (const float*)d_in[2];
  const float* Wq   = (const float*)d_in[3];
  const float* Wk   = (const float*)d_in[4];
  const float* Wv   = (const float*)d_in[5];
  const float* Wo   = (const float*)d_in[6];
  const float* bo   = (const float*)d_in[7];
  float* out = (float*)d_out;

  char* ws = (char*)d_ws;
  size_t off = 0;
  auto carve = [&](size_t elems) {
    _Float16* p = (_Float16*)(ws + off);
    off += ((elems * sizeof(_Float16) + 255) & ~(size_t)255);
    return p;
  };
  _Float16* xn   = carve((size_t)MROWS * EMBED);   // normalized activations
  _Float16* wq16 = carve((size_t)EMBED * EMBED);
  _Float16* wk16 = carve((size_t)EMBED * EMBED);
  _Float16* wv16 = carve((size_t)EMBED * EMBED);
  _Float16* wo16 = carve((size_t)EMBED * EMBED);
  _Float16* qh   = carve((size_t)MROWS * EMBED);   // [B,H,N,D]
  _Float16* kh   = carve((size_t)MROWS * EMBED);   // [B,H,N,D]
  _Float16* vth  = carve((size_t)MROWS * EMBED);   // [B,H,D,N]
  _Float16* ao   = carve((size_t)MROWS * EMBED);   // attention output [B*N, E]

  int wcnt = EMBED * EMBED;
  dim3 cvtGrid((wcnt + 255) / 256);
  cvt_f32_f16<<<cvtGrid, 256, 0, stream>>>(Wq, wq16, wcnt);
  cvt_f32_f16<<<cvtGrid, 256, 0, stream>>>(Wk, wk16, wcnt);
  cvt_f32_f16<<<cvtGrid, 256, 0, stream>>>(Wv, wv16, wcnt);
  cvt_f32_f16<<<cvtGrid, 256, 0, stream>>>(Wo, wo16, wcnt);

  ln_kernel<<<dim3(MROWS), 32, 0, stream>>>(x, ln_w, ln_b, xn);

  dim3 gemmGrid((MROWS / 64) * (EMBED / 64));     // 4 waves/block, 16x64 each
  gemm_qkv<<<gemmGrid, 128, 0, stream>>>(xn, wq16, qh, 1);
  gemm_qkv<<<gemmGrid, 128, 0, stream>>>(xn, wk16, kh, 1);
  gemm_qkv<<<gemmGrid, 128, 0, stream>>>(xn, wv16, vth, 2);

  flash_attn<<<dim3(BATCH * HEADS * (NSEQ / 16) / 4), 128, 0, stream>>>(qh, kh, vth, ao);

  out_proj<<<gemmGrid, 128, 0, stream>>>(ao, wo16, x, bo, out);
}
